// MoleculeGNN_3942779978279
// MI455X (gfx1250) — matrix operand order repaired and checked by reading, not compiled, not run on hardware
//
#include <hip/hip_runtime.h>
#include <hip/hip_bf16.h>

// ---------------------------------------------------------------------------
// 2-layer GCN + global mean pool for MI455X (gfx1250, wave32).
// GEMMs use v_wmma_f32_16x16x32_bf16 with a bf16 hi/lo split (3 WMMAs) for
// near-fp32 accuracy. Edge aggregation is wave-per-edge float4 atomics (L2).
// ---------------------------------------------------------------------------

typedef __attribute__((ext_vector_type(16))) __bf16         v16bf;
typedef __attribute__((ext_vector_type(8)))  float          v8f;
typedef __attribute__((ext_vector_type(16))) unsigned short us16;

__device__ __forceinline__ unsigned short f2bf(float f) {
    unsigned u = __builtin_bit_cast(unsigned, f);
    u += 0x7fffu + ((u >> 16) & 1u);          // round to nearest even
    return (unsigned short)(u >> 16);
}
__device__ __forceinline__ float bf2f(unsigned short h) {
    return __builtin_bit_cast(float, (unsigned)h << 16);
}

// Pack W[128][128] (row-major, K x N) into WMMA B-fragment order:
// frag(kt,nt): lane l = (K>>4 & 1)*16 + (N&15), element i = K&15.
__global__ void gcn_pack_w(const float* __restrict__ W,
                           unsigned short* __restrict__ hi,
                           unsigned short* __restrict__ lo) {
    int t = blockIdx.x * blockDim.x + threadIdx.x;
    if (t >= 128 * 128) return;
    int K = t >> 7, Ncol = t & 127;
    float w = W[t];
    unsigned short h = f2bf(w);
    unsigned short l = f2bf(w - bf2f(h));
    int kt = K >> 5, halfk = (K >> 4) & 1, i = K & 15;
    int nt = Ncol >> 4, lane = halfk * 16 + (Ncol & 15);
    int idx = (((kt * 8 + nt) * 32) + lane) * 16 + i;
    hi[idx] = h;
    lo[idx] = l;
}

__global__ void gcn_deg_init(float* __restrict__ deg, int N) {
    int i = blockIdx.x * blockDim.x + threadIdx.x;
    if (i < N) deg[i] = 1.0f;                 // self-loop
}
__global__ void gcn_deg_edge(const long long* __restrict__ ei,
                             float* __restrict__ deg, int E) {
    int i = blockIdx.x * blockDim.x + threadIdx.x;
    if (i < E) atomicAdd(&deg[(int)ei[(size_t)E + i]], 1.0f);
}
__global__ void gcn_deg_rsqrt(float* __restrict__ deg, int N) {
    int i = blockIdx.x * blockDim.x + threadIdx.x;
    if (i < N) deg[i] = rsqrtf(deg[i]);
}

__global__ void gcn_zero(float* __restrict__ p, int n) {
    int i = blockIdx.x * blockDim.x + threadIdx.x;
    if (i < n) p[i] = 0.0f;
}

// One wave per 16-row tile; full 128-column sweep (8 n-tiles).
// Gout[row][col] = dis[row] * (A @ W)[row][col]
__global__ void gcn_gemm_g(const float* __restrict__ A,
                           const unsigned short* __restrict__ Bh,
                           const unsigned short* __restrict__ Bl,
                           const float* __restrict__ dis,
                           float* __restrict__ Gout, int ntiles) {
    int wave = (blockIdx.x * blockDim.x + threadIdx.x) >> 5;
    if (wave >= ntiles) return;               // uniform per wave
    int lane = threadIdx.x & 31;
    int r = lane & 15, half = lane >> 4;
    int row0 = wave * 16;

    v8f acc[8] = {};
    const float* arow = A + (size_t)(row0 + r) * 128;

#pragma unroll
    for (int kt = 0; kt < 4; ++kt) {
        // Build A fragment (lanes 0-15 K 0-15, lanes 16-31 K 16-31,
        // contiguous to match packed B) with hi/lo bf16 split.
        us16 ah, al;
#pragma unroll
        for (int grp = 0; grp < 2; ++grp) {
            int kb = kt * 32 + half * 16 + grp * 8;
            float4 f0 = *(const float4*)(arow + kb);
            float4 f1 = *(const float4*)(arow + kb + 4);
            float fv[8] = {f0.x, f0.y, f0.z, f0.w, f1.x, f1.y, f1.z, f1.w};
#pragma unroll
            for (int j = 0; j < 8; ++j) {
                unsigned short h = f2bf(fv[j]);
                ah[grp * 8 + j] = h;
                al[grp * 8 + j] = f2bf(fv[j] - bf2f(h));
            }
        }
        v16bf Ah = __builtin_bit_cast(v16bf, ah);
        v16bf Al = __builtin_bit_cast(v16bf, al);

#pragma unroll
        for (int nt = 0; nt < 8; ++nt) {
            size_t fo = (((size_t)(kt * 8 + nt) * 32 + lane) << 4);
            v16bf Bhv = __builtin_bit_cast(v16bf, *(const us16*)(Bh + fo));
            v16bf Blv = __builtin_bit_cast(v16bf, *(const us16*)(Bl + fo));
            acc[nt] = __builtin_amdgcn_wmma_f32_16x16x32_bf16(
                false, Ah, false, Bhv, (short)0, acc[nt], false, false);
            acc[nt] = __builtin_amdgcn_wmma_f32_16x16x32_bf16(
                false, Ah, false, Blv, (short)0, acc[nt], false, false);
            acc[nt] = __builtin_amdgcn_wmma_f32_16x16x32_bf16(
                false, Al, false, Bhv, (short)0, acc[nt], false, false);
        }
    }

    // C/D layout: VGPR i -> M = i + 8*(lane/16), N = lane%16 (+16*nt)
    float dv[8];
#pragma unroll
    for (int i = 0; i < 8; ++i) dv[i] = dis[row0 + half * 8 + i];
#pragma unroll
    for (int nt = 0; nt < 8; ++nt) {
        int col = nt * 16 + r;
#pragma unroll
        for (int i = 0; i < 8; ++i)
            Gout[(size_t)(row0 + half * 8 + i) * 128 + col] = dv[i] * acc[nt][i];
    }
}

// Wave per edge: acc[dst] += g[src]  (128 floats, float4/lane, L2 atomics)
__global__ void gcn_scatter(const long long* __restrict__ ei,
                            const float* __restrict__ g,
                            float* __restrict__ acc, int E) {
    int wave = (blockIdx.x * blockDim.x + threadIdx.x) >> 5;
    if (wave >= E) return;
    int lane = threadIdx.x & 31;
    int s = (int)ei[wave];
    int d = (int)ei[(size_t)E + wave];
    float4 v = *(const float4*)(g + (size_t)s * 128 + lane * 4);
    float* p = acc + (size_t)d * 128 + lane * 4;
    atomicAdd(p + 0, v.x);
    atomicAdd(p + 1, v.y);
    atomicAdd(p + 2, v.z);
    atomicAdd(p + 3, v.w);
}

// In-place: g[v][c] = relu(dis[v]*(acc[v][c] + g[v][c]) + b[c])
__global__ void gcn_finalize(float* __restrict__ g, const float* __restrict__ acc,
                             const float* __restrict__ dis,
                             const float* __restrict__ bias, int N) {
    int idx4 = blockIdx.x * blockDim.x + threadIdx.x;
    if (idx4 >= N * 32) return;
    int v = idx4 >> 5;
    int c = (idx4 & 31) * 4;
    float4 gv = *(const float4*)(g + (size_t)idx4 * 4);
    float4 av = *(const float4*)(acc + (size_t)idx4 * 4);
    float4 bv = *(const float4*)(bias + c);
    float dv = dis[v];
    float4 o;
    o.x = fmaxf(dv * (gv.x + av.x) + bv.x, 0.0f);
    o.y = fmaxf(dv * (gv.y + av.y) + bv.y, 0.0f);
    o.z = fmaxf(dv * (gv.z + av.z) + bv.z, 0.0f);
    o.w = fmaxf(dv * (gv.w + av.w) + bv.w, 0.0f);
    *(float4*)(g + (size_t)idx4 * 4) = o;
}

__global__ void gcn_pool_acc(const float* __restrict__ h,
                             const long long* __restrict__ batch,
                             float* __restrict__ pool, float* __restrict__ cnt,
                             int N) {
    int wave = (blockIdx.x * blockDim.x + threadIdx.x) >> 5;
    if (wave >= N) return;
    int lane = threadIdx.x & 31;
    int b = (int)batch[wave];
    float4 v = *(const float4*)(h + (size_t)wave * 128 + lane * 4);
    float* p = pool + (size_t)b * 128 + lane * 4;
    atomicAdd(p + 0, v.x);
    atomicAdd(p + 1, v.y);
    atomicAdd(p + 2, v.z);
    atomicAdd(p + 3, v.w);
    if (lane == 0) atomicAdd(cnt + b, 1.0f);
}

__global__ void gcn_pool_div(const float* __restrict__ pool,
                             const float* __restrict__ cnt,
                             float* __restrict__ out, int G) {
    int idx = blockIdx.x * blockDim.x + threadIdx.x;
    if (idx >= G * 128) return;
    out[idx] = pool[idx] / fmaxf(cnt[idx >> 7], 1.0f);
}

extern "C" void kernel_launch(void* const* d_in, const int* in_sizes, int n_in,
                              void* d_out, int out_size, void* d_ws, size_t ws_size,
                              hipStream_t stream) {
    (void)n_in; (void)ws_size;
    const float*     x     = (const float*)d_in[0];
    const long long* ei    = (const long long*)d_in[1];
    const long long* batch = (const long long*)d_in[2];
    const float*     W1    = (const float*)d_in[4];
    const float*     b1    = (const float*)d_in[5];
    const float*     W2    = (const float*)d_in[6];
    const float*     b2    = (const float*)d_in[7];
    float*           out   = (float*)d_out;

    const int N = in_sizes[0] / 128;   // 50000 (multiple of 16)
    const int E = in_sizes[1] / 2;     // 800000
    const int G = out_size / 128;      // 500

    // workspace carve-out (256B aligned)
    char* ws = (char*)d_ws;
    size_t off = 0;
    auto take = [&](size_t bytes) -> char* {
        char* p = ws + off;
        off = (off + bytes + 255) & ~(size_t)255;
        return p;
    };
    float* dis          = (float*)take((size_t)N * 4);
    float* P            = (float*)take((size_t)N * 512);
    float* Q            = (float*)take((size_t)N * 512);
    unsigned short* w1h = (unsigned short*)take(32768);
    unsigned short* w1l = (unsigned short*)take(32768);
    unsigned short* w2h = (unsigned short*)take(32768);
    unsigned short* w2l = (unsigned short*)take(32768);
    float* pool         = (float*)take((size_t)G * 512);
    float* cnt          = (float*)take((size_t)G * 4);

    const int BT = 256;
    const int ntiles     = N / 16;
    const int gemmBlocks = (ntiles * 32 + BT - 1) / BT;
    const int scatBlocks = (E * 32 + BT - 1) / BT;
    const int finBlocks  = (N * 32 + BT - 1) / BT;
    const int zeroBlocks = (N * 128 + BT - 1) / BT;

    // setup: W fragments + symmetric normalization
    gcn_pack_w<<<64, BT, 0, stream>>>(W1, w1h, w1l);
    gcn_pack_w<<<64, BT, 0, stream>>>(W2, w2h, w2l);
    gcn_deg_init<<<(N + BT - 1) / BT, BT, 0, stream>>>(dis, N);
    gcn_deg_edge<<<(E + BT - 1) / BT, BT, 0, stream>>>(ei, dis, E);
    gcn_deg_rsqrt<<<(N + BT - 1) / BT, BT, 0, stream>>>(dis, N);

    // layer 1: g1 = dis*(x@W1) -> P ; acc -> Q ; h1 (in-place) -> P
    gcn_gemm_g<<<gemmBlocks, BT, 0, stream>>>(x, w1h, w1l, dis, P, ntiles);
    gcn_zero<<<zeroBlocks, BT, 0, stream>>>(Q, N * 128);
    gcn_scatter<<<scatBlocks, BT, 0, stream>>>(ei, P, Q, E);
    gcn_finalize<<<finBlocks, BT, 0, stream>>>(P, Q, dis, b1, N);

    // layer 2: g2 = dis*(h1@W2) -> Q ; acc -> P ; h2 (in-place) -> Q
    gcn_gemm_g<<<gemmBlocks, BT, 0, stream>>>(P, w2h, w2l, dis, Q, ntiles);
    gcn_zero<<<zeroBlocks, BT, 0, stream>>>(P, N * 128);
    gcn_scatter<<<scatBlocks, BT, 0, stream>>>(ei, Q, P, E);
    gcn_finalize<<<finBlocks, BT, 0, stream>>>(Q, P, dis, b2, N);

    // global mean pool
    gcn_zero<<<(G * 128 + BT - 1) / BT, BT, 0, stream>>>(pool, G * 128);
    gcn_zero<<<(G + BT - 1) / BT, BT, 0, stream>>>(cnt, G);
    gcn_pool_acc<<<(N * 32 + BT - 1) / BT, BT, 0, stream>>>(Q, batch, pool, cnt, N);
    gcn_pool_div<<<(G * 128 + BT - 1) / BT, BT, 0, stream>>>(pool, cnt, out, G);
}
